// conv_interp_5583457484820
// MI455X (gfx1250) — matrix operand order repaired and checked by reading, not compiled, not run on hardware
//
#include <hip/hip_runtime.h>
#include <hip/hip_bf16.h>

typedef __attribute__((ext_vector_type(16))) _Float16 v16h;
typedef __attribute__((ext_vector_type(8)))  float    v8f;
typedef __attribute__((ext_vector_type(4)))  int      v4i;

#if defined(__has_builtin)
#if __has_builtin(__builtin_amdgcn_global_load_async_to_lds_b128) && __has_builtin(__builtin_amdgcn_s_wait_asynccnt)
#define HAVE_ASYNC 1
#endif
#endif
#ifndef HAVE_ASYNC
#define HAVE_ASYNC 0
#endif

// ---------------------------------------------------------------------------
// Kernel 1: interpolate weights per batch, emit f16 pre-swizzled into the
// WMMA A-fragment layout (16-bit A 16x32, ISA 7.12.2):
//   lane = khalf*16 + r  (r = row M within 16-row tile)
//   record[16 f16]: j<8 -> Klocal = khalf*8 + j ; j>=8 -> Klocal = 16 + khalf*8 + (j-8)
// K ordering: k = (kh*3+kw)*64 + ci   (18 chunks of 32)
// Record index: ((b*4 + mt)*18 + kc)*32 + lane
// ---------------------------------------------------------------------------
__global__ __launch_bounds__(256) void pack_w_kernel(
    const float* __restrict__ W0, const float* __restrict__ W1,
    const float* __restrict__ DoT, _Float16* __restrict__ out) {
  int idx = blockIdx.x * 256 + threadIdx.x;            // 0 .. 36863
  if (idx >= 16 * 4 * 18 * 32) return;
  const int lane = idx & 31;
  const int kc   = (idx >> 5) % 18;
  const int mt   = (idx / (32 * 18)) & 3;
  const int b    = idx / (32 * 18 * 4);
  const float d  = DoT[b];
  const int khalf = lane >> 4, r = lane & 15;
  const int m = mt * 16 + r;
  _Float16* o = out + (size_t)idx * 16;
#pragma unroll
  for (int j = 0; j < 16; ++j) {
    const int klocal = (j < 8) ? (khalf * 8 + j) : (16 + khalf * 8 + (j - 8));
    const int k   = kc * 32 + klocal;
    const int tap = k >> 6, ci = k & 63;
    const int kh = tap / 3, kw = tap % 3;
    const int wi = ((m * 64 + ci) * 3 + kh) * 3 + kw;    // OIHW
    const float v = (1.0f - d) * W0[wi] + d * W1[wi];
    o[j] = (_Float16)v;
  }
}

// ---------------------------------------------------------------------------
// Kernel 2: implicit-GEMM conv. Block (256 thr, 8 waves) -> one batch b,
// 4 output rows x 16 cols x all 64 Cout. Wave (mt,nt): Cout tile mt*16..+15,
// rows {2nt, 2nt+1}. K = 576 = 3x3 taps * 64 Cin, chunked by 32.
// ---------------------------------------------------------------------------
union V16U { uint4 q[2]; v16h v; };

__global__ __launch_bounds__(256) void conv_wmma_kernel(
    const float* __restrict__ x, const _Float16* __restrict__ Wp,
    float* __restrict__ y) {
  __shared__ __align__(16) _Float16 sW[4 * 18 * 32 * 16];  // 73728 B packed A frags
  __shared__ __align__(16) _Float16 sX[6 * 18 * 72];       // 15552 B patch [row][col][ci], 72-pad

  const int tid  = threadIdx.x;
  const int lane = tid & 31, wv = tid >> 5;
  const int mt = wv >> 1, nt = wv & 1;
  const int b = blockIdx.z, h0 = blockIdx.y * 4, w0 = blockIdx.x * 16;

  // ---- stage packed weights for this batch: 73728 B = 4608 x b128 ----
  const uint4* gq = (const uint4*)(Wp + (size_t)b * (4 * 18 * 32 * 16));
  uint4* sq = (uint4*)sW;
#if HAVE_ASYNC
#pragma unroll 2
  for (int i = tid; i < 4608; i += 256) {
    __builtin_amdgcn_global_load_async_to_lds_b128(
        (__attribute__((address_space(1))) v4i*)(uintptr_t)(gq + i),
        (__attribute__((address_space(3))) v4i*)(unsigned)(uintptr_t)(sq + i),
        0, 0);
  }
#else
#pragma unroll 2
  for (int i = tid; i < 4608; i += 256) sq[i] = gq[i];
#endif

  // ---- stage input patch (6 rows x 18 cols x 64 ci), f32 -> f16, zero pad ----
  for (int i = tid; i < 6912; i += 256) {
    const int ci = i / 108, rem = i % 108;
    const int row = rem / 18, col = rem % 18;
    const int h = h0 - 1 + row, w = w0 - 1 + col;
    float v = 0.0f;
    if ((unsigned)h < 192u && (unsigned)w < 192u)
      v = x[(((size_t)b * 64 + ci) * 192 + h) * 192 + w];
    sX[(row * 18 + col) * 72 + ci] = (_Float16)v;
  }
#if HAVE_ASYNC
  __builtin_amdgcn_s_wait_asynccnt(0);
#endif
  __syncthreads();

  const int khalf = lane >> 4, nn = lane & 15;
  v8f acc0 = {0.f, 0.f, 0.f, 0.f, 0.f, 0.f, 0.f, 0.f};
  v8f acc1 = {0.f, 0.f, 0.f, 0.f, 0.f, 0.f, 0.f, 0.f};

  // ---- K loop: 3x3 taps x 2 ci-halves = 18 chunks of K=32, 2 WMMAs each ----
#pragma unroll
  for (int kh = 0; kh < 3; ++kh) {
#pragma unroll
    for (int kw = 0; kw < 3; ++kw) {
#pragma unroll
      for (int cih = 0; cih < 2; ++cih) {
        const int kc = (kh * 3 + kw) * 2 + cih;
        // A fragment: one 32B record per lane (pre-swizzled)
        const _Float16* ap = sW + (((mt * 18 + kc) * 32 + lane) << 4);
        V16U ua;
        ua.q[0] = *(const uint4*)ap;
        ua.q[1] = *(const uint4*)(ap + 8);
        // B fragment: lane half khalf holds K rows khalf*8+(0..7), 16+khalf*8+(0..7);
        // lane's column = pixel nn; K row -> ci = cih*32 + K row
        const int ci0 = cih * 32 + khalf * 8;
#pragma unroll
        for (int s = 0; s < 2; ++s) {
          const int r = nt * 2 + s;
          const _Float16* bp = sX + ((r + kh) * 18 + (nn + kw)) * 72 + ci0;
          V16U ub;
          ub.q[0] = *(const uint4*)bp;
          ub.q[1] = *(const uint4*)(bp + 16);
          v8f& acc = s ? acc1 : acc0;
          acc = __builtin_amdgcn_wmma_f32_16x16x32_f16(
              false, ua.v, false, ub.v, (short)0, acc, false, false);
        }
      }
    }
  }

  // ---- store: C/D layout: lanes 0-15 -> M=j, lanes 16-31 -> M=j+8; N=lane&15
#pragma unroll
  for (int s = 0; s < 2; ++s) {
    const v8f acc = s ? acc1 : acc0;
    const int h = h0 + nt * 2 + s;
#pragma unroll
    for (int j = 0; j < 8; ++j) {
      const int m = mt * 16 + khalf * 8 + j;
      y[(((size_t)b * 64 + m) * 192 + h) * 192 + (w0 + nn)] = acc[j];
    }
  }
}

// ---------------------------------------------------------------------------
// Launcher. Inputs (setup_inputs order): x, DoT, W0, W1. Workspace: 1.15 MB
// packed f16 weights.
// ---------------------------------------------------------------------------
extern "C" void kernel_launch(void* const* d_in, const int* in_sizes, int n_in,
                              void* d_out, int out_size, void* d_ws, size_t ws_size,
                              hipStream_t stream) {
  const float* x   = (const float*)d_in[0];
  const float* DoT = (const float*)d_in[1];
  const float* W0  = (const float*)d_in[2];
  const float* W1  = (const float*)d_in[3];
  float* y = (float*)d_out;
  _Float16* Wp = (_Float16*)d_ws;  // needs 16*4*18*32*16*2 = 1,179,648 B

  pack_w_kernel<<<144, 256, 0, stream>>>(W0, W1, DoT, Wp);
  dim3 grid(192 / 16, 192 / 4, 16);  // (12, 48, 16)
  conv_wmma_kernel<<<grid, 256, 0, stream>>>(x, Wp, y);
}
